// Attention_63221918597766
// MI455X (gfx1250) — matrix operand order repaired and checked by reading, not compiled, not run on hardware
//
#include <hip/hip_runtime.h>
#include <hip/hip_bf16.h>

// MI455X (gfx1250) channel-attention pipeline.
// - activations in bf16 (HBM 23.3 TB/s bound everywhere except GEMM1)
// - all matrix math: v_wmma_f32_16x16x32_bf16, f32 accumulate
// - GEMM A-tiles staged via GLOBAL_LOAD_ASYNC_TO_LDS_B128 (ASYNCcnt,
//   double-buffered, s_wait_asynccnt pipelining)
// - attn@v + 1x1 proj fused: W_eff = W_proj @ blockdiag(attn)

typedef __attribute__((ext_vector_type(16))) __bf16 v16bf;
typedef __attribute__((ext_vector_type(8)))  float  v8f;

#define B_    2
#define C_    384
#define HEADS 8
#define HD    48
#define H_    256
#define W_    256
#define N_    65536L
#define OC3   1152

union Frag16 { v16bf v; uint4 q[2]; };

// A-matrix 16x32 bf16 frag (ISA 7.12.2): lane row = lane&15,
// K chunks of 8 at (lane>>4)*8 and (lane>>4)*8 + 16.
__device__ __forceinline__ v16bf load_a_frag_lds(const __bf16* base, int stride, int lane) {
  int m  = lane & 15;
  int ko = (lane >> 4) * 8;
  Frag16 f;
  const uint4* p = (const uint4*)(base + m * stride + ko);
  f.q[0] = p[0];
  f.q[1] = p[2];      // +16 elements = +32 bytes
  return f.v;
}

// B-matrix 32x16 bf16 frag: lane col = lane&15,
// K chunks of 8 at (lane>>4)*8 and (lane>>4)*8 + 16.
__device__ __forceinline__ v16bf load_b_frag_lds(const __bf16* base, int stride, int lane) {
  int n  = lane & 15;
  int ko = (lane >> 4) * 8;
  Frag16 f;
  const uint4* p = (const uint4*)(base + n * stride + ko);
  f.q[0] = p[0];
  f.q[1] = p[2];
  return f.v;
}

// One 16-byte async copy global->LDS per lane (ASYNCcnt-tracked).
__device__ __forceinline__ void async_copy_b128(const void* gptr, void* lds_generic) {
  // flat LDS aperture: addr[31:0] is the LDS byte offset (ISA 10.2)
  uint32_t lds = (uint32_t)(uintptr_t)lds_generic;
  uint64_t ga  = (uint64_t)(uintptr_t)gptr;
  asm volatile("global_load_async_to_lds_b128 %0, %1, off"
               :: "v"(lds), "v"(ga) : "memory");
}

// ---------------------------------------------------------------------------
// bf16 WMMA GEMM: C[b] = A[b](MxK) * B[b](KxN). 256 thr = 8 waves,
// tile 128x128x32; wave owns 32x64 (2x4 accum tiles). A staged async +
// double buffered; B staged transposed (scalar) so frags are 16B LDS reads.
// ---------------------------------------------------------------------------
template<typename OT>
__global__ __launch_bounds__(256)
void gemm_wmma(const __bf16* __restrict__ A, const __bf16* __restrict__ Bm,
               OT* __restrict__ Cm, int M, int K, long Ncols,
               long sA, long sB, long sC)
{
  constexpr int BM = 128, BN = 128, BK = 32, LD = 40;  // 80B rows: 16B aligned
  __shared__ __align__(16) __bf16 As[2][BM * LD];
  __shared__ __align__(16) __bf16 Bs[2][BN * LD];

  const __bf16* Ab = A  + (long)blockIdx.z * sA;
  const __bf16* Bb = Bm + (long)blockIdx.z * sB;
  OT*           Cb = Cm + (long)blockIdx.z * sC;

  int  m0 = blockIdx.y * BM;
  long n0 = (long)blockIdx.x * BN;
  int tid = threadIdx.x, lane = tid & 31, wid = tid >> 5;
  int wm = (wid >> 1) * 32;
  int wn = (wid & 1)  * 64;

  // A tile: 128 rows x 32 bf16 = 512 chunks of 16B; 2 async issues per thread.
  auto stageA = [&](int buf, int k0) {
    #pragma unroll
    for (int i = 0; i < 2; ++i) {
      int c = tid + i * 256;
      int r = c >> 2, q = (c & 3) * 8;
      async_copy_b128(Ab + (long)(m0 + r) * K + k0 + q, &As[buf][r * LD + q]);
    }
  };
  // B tile 32x128 -> LDS transposed Bs[n][k] (scalar: transpose in the write)
  auto stageB = [&](int buf, int k0) {
    int kk = tid >> 3, no = (tid & 7) * 16;
    const __bf16* src = Bb + (long)(k0 + kk) * Ncols + n0 + no;
    __builtin_prefetch(src + (long)BK * Ncols, 0, 1);
    #pragma unroll
    for (int i = 0; i < 16; ++i) Bs[buf][(no + i) * LD + kk] = src[i];
  };

  v8f acc[2][4] = {};
  stageA(0, 0);
  stageB(0, 0);

  int nk = K / BK;
  for (int t = 0; t < nk; ++t) {
    int buf = t & 1;
    if (t + 1 < nk) {
      stageA(buf ^ 1, (t + 1) * BK);
      stageB(buf ^ 1, (t + 1) * BK);
      // keep only the newest 2 async copies in flight -> tile t is resident
      asm volatile("s_wait_asynccnt 0x2" ::: "memory");
    } else {
      asm volatile("s_wait_asynccnt 0x0" ::: "memory");
    }
    __syncthreads();
    #pragma unroll
    for (int i = 0; i < 2; ++i) {
      v16bf a = load_a_frag_lds(&As[buf][(wm + i * 16) * LD], LD, lane);
      #pragma unroll
      for (int j = 0; j < 4; ++j) {
        v16bf b = load_b_frag_lds(&Bs[buf][(wn + j * 16) * LD], LD, lane);
        acc[i][j] = __builtin_amdgcn_wmma_f32_16x16x32_bf16(
            false, a, false, b, (short)0, acc[i][j], false, false);
      }
    }
    __syncthreads();
  }

  #pragma unroll
  for (int i = 0; i < 2; ++i)
    #pragma unroll
    for (int j = 0; j < 4; ++j)
      #pragma unroll
      for (int r = 0; r < 8; ++r) {
        int  row = m0 + wm + i * 16 + r + (lane >> 4) * 8;
        long col = n0 + wn + j * 16 + (lane & 15);
        Cb[(long)row * Ncols + col] = (OT)acc[i][j][r];
      }
}

// ---------------------------------------------------------------------------
// f32 -> bf16 bulk convert (vectorized): n4 = #float4 groups.
// ---------------------------------------------------------------------------
__global__ __launch_bounds__(256)
void cvt_f32_bf16(const float* __restrict__ in, __bf16* __restrict__ out, long n4)
{
  long i = (long)blockIdx.x * 256 + threadIdx.x;
  if (i >= n4) return;
  float4 v = ((const float4*)in)[i];
  __bf16 o[4] = { (__bf16)v.x, (__bf16)v.y, (__bf16)v.z, (__bf16)v.w };
  *(uint2*)(out + i * 4) = *(const uint2*)o;
}

// ---------------------------------------------------------------------------
// Depthwise 3x3, SAME padding. One block per (b, c, y); 256 threads = W.
// ---------------------------------------------------------------------------
__global__ __launch_bounds__(256)
void dwconv3x3(const __bf16* __restrict__ in, const float* __restrict__ wdw,
               __bf16* __restrict__ out)
{
  int b = blockIdx.z, c = blockIdx.y, y = blockIdx.x;
  __shared__ __bf16 rows[3][W_ + 2];
  const __bf16* base = in + ((long)b * OC3 + c) * N_;
  int x = threadIdx.x;
  #pragma unroll
  for (int dy = 0; dy < 3; ++dy) {
    int yy = y + dy - 1;
    __bf16 v = (__bf16)0.0f;
    if (yy >= 0 && yy < H_) v = base[(long)yy * W_ + x];
    rows[dy][x + 1] = v;
    if (x == 0) { rows[dy][0] = (__bf16)0.0f; rows[dy][W_ + 1] = (__bf16)0.0f; }
  }
  __syncthreads();
  float w[9];
  #pragma unroll
  for (int i = 0; i < 9; ++i) w[i] = wdw[c * 9 + i];
  float s = 0.f;
  #pragma unroll
  for (int dy = 0; dy < 3; ++dy)
    #pragma unroll
    for (int dx = 0; dx < 3; ++dx)
      s += w[dy * 3 + dx] * (float)rows[dy][x + dx];
  out[((long)b * OC3 + c) * N_ + (long)y * W_ + x] = (__bf16)s;
}

// ---------------------------------------------------------------------------
// Inverse L2 norms of the 2*768 q/k rows (length 65536).
// ---------------------------------------------------------------------------
__global__ __launch_bounds__(256)
void rownorms(const __bf16* __restrict__ qk, float* __restrict__ inv)
{
  int r = blockIdx.x;               // 0..1535
  int which = r / 768;              // 0 = q, 1 = k
  int rr = r % 768;
  int b = rr / 384, ch = rr % 384;
  const __bf16* row = qk + ((long)b * OC3 + which * C_ + ch) * N_;
  const uint4* p = (const uint4*)row;
  float s = 0.f;
  for (int i = threadIdx.x; i < (int)(N_ / 8); i += 256) {
    uint4 u = p[i];
    const __bf16* e = (const __bf16*)&u;
    #pragma unroll
    for (int j = 0; j < 8; ++j) { float v = (float)e[j]; s += v * v; }
  }
  __shared__ float red[256];
  red[threadIdx.x] = s;
  __syncthreads();
  for (int off = 128; off > 0; off >>= 1) {
    if (threadIdx.x < off) red[threadIdx.x] += red[threadIdx.x + off];
    __syncthreads();
  }
  if (threadIdx.x == 0) inv[r] = 1.0f / fmaxf(sqrtf(red[0]), 1e-12f);
}

// ---------------------------------------------------------------------------
// attn[b,h] = softmax( (q @ k^T) * invq * invk * tau ). One block per (b,h):
// 8 waves split K=65536, each accumulating all 3x3 WMMA tiles, LDS reduce.
// ---------------------------------------------------------------------------
__global__ __launch_bounds__(256)
void chan_attn(const __bf16* __restrict__ qkv, const float* __restrict__ inv,
               const float* __restrict__ temp, float* __restrict__ attn)
{
  int bh = blockIdx.x;            // 0..15
  int b = bh >> 3, hh = bh & 7;
  int tid = threadIdx.x, lane = tid & 31, wid = tid >> 5;
  const __bf16* qbase = qkv + ((long)b * OC3 + hh * HD) * N_;
  const __bf16* kbase = qkv + ((long)b * OC3 + C_ + hh * HD) * N_;

  v8f acc[3][3] = {};
  for (int step = 0; step < 256; ++step) {       // 2048 K-chunks / 8 waves
    long n0 = ((long)(step * 8 + wid)) * 32;
    v16bf af[3], bf[3];
    #pragma unroll
    for (int i = 0; i < 3; ++i) {                // q rows, A interleaved layout
      const __bf16* rp = qbase + (long)(i * 16 + (lane & 15)) * N_ + n0 + (lane >> 4) * 8;
      Frag16 f; const uint4* p = (const uint4*)rp;
      f.q[0] = p[0]; f.q[1] = p[2];
      af[i] = f.v;
    }
    #pragma unroll
    for (int j = 0; j < 3; ++j) {                // k rows as B cols, interleaved
      const __bf16* rp = kbase + (long)(j * 16 + (lane & 15)) * N_ + n0 + (lane >> 4) * 8;
      Frag16 f; const uint4* p = (const uint4*)rp;
      f.q[0] = p[0]; f.q[1] = p[2];
      bf[j] = f.v;
    }
    #pragma unroll
    for (int i = 0; i < 3; ++i)
      #pragma unroll
      for (int j = 0; j < 3; ++j)
        acc[i][j] = __builtin_amdgcn_wmma_f32_16x16x32_bf16(
            false, af[i], false, bf[j], (short)0, acc[i][j], false, false);
  }

  __shared__ float red[8][HD * HD];
  #pragma unroll
  for (int i = 0; i < 3; ++i)
    #pragma unroll
    for (int j = 0; j < 3; ++j)
      #pragma unroll
      for (int r = 0; r < 8; ++r) {
        int row = i * 16 + r + (lane >> 4) * 8;
        int col = j * 16 + (lane & 15);
        red[wid][row * HD + col] = acc[i][j][r];
      }
  __syncthreads();

  __shared__ float S[HD * HD];
  float t = temp[hh];
  for (int idx = tid; idx < HD * HD; idx += 256) {
    float s = 0.f;
    #pragma unroll
    for (int wv = 0; wv < 8; ++wv) s += red[wv][idx];
    int cc = idx / HD, dd = idx % HD;
    float iq = inv[b * 384 + hh * HD + cc];
    float ik = inv[768 + b * 384 + hh * HD + dd];
    S[idx] = s * iq * ik * t;
  }
  __syncthreads();

  if (tid < HD) {
    int cc = tid;
    float mx = -1e30f;
    for (int d = 0; d < HD; ++d) mx = fmaxf(mx, S[cc * HD + d]);
    float e[HD], ssum = 0.f;
    for (int d = 0; d < HD; ++d) { e[d] = __expf(S[cc * HD + d] - mx); ssum += e[d]; }
    float rs = 1.0f / ssum;
    float* dst = attn + ((long)bh * HD + cc) * HD;
    for (int d = 0; d < HD; ++d) dst[d] = e[d] * rs;
  }
}

// ---------------------------------------------------------------------------
// W_eff[b] = W_proj @ blockdiag(attn[b])  (384x384, bf16 out).
// ---------------------------------------------------------------------------
__global__ __launch_bounds__(384)
void build_weff(const float* __restrict__ wproj, const float* __restrict__ attn,
                __bf16* __restrict__ weff)
{
  int b = blockIdx.y, o = blockIdx.x;
  __shared__ float at[HEADS * HD * HD];
  for (int i = threadIdx.x; i < HEADS * HD * HD; i += 384)
    at[i] = attn[(long)b * HEADS * HD * HD + i];
  __shared__ float wrow[C_];
  for (int i = threadIdx.x; i < C_; i += 384) wrow[i] = wproj[(long)o * C_ + i];
  __syncthreads();
  int j = threadIdx.x;
  int hh = j / HD, d = j % HD;
  float s = 0.f;
  #pragma unroll
  for (int c = 0; c < HD; ++c)
    s += wrow[hh * HD + c] * at[(hh * HD + c) * HD + d];
  weff[((long)b * C_ + o) * C_ + j] = (__bf16)s;
}

// ---------------------------------------------------------------------------
extern "C" void kernel_launch(void* const* d_in, const int* in_sizes, int n_in,
                              void* d_out, int out_size, void* d_ws, size_t ws_size,
                              hipStream_t stream) {
  const float* x           = (const float*)d_in[0];
  const float* w_qkv       = (const float*)d_in[1];
  const float* w_dw        = (const float*)d_in[2];
  const float* w_proj      = (const float*)d_in[3];
  const float* temperature = (const float*)d_in[4];
  float* out = (float*)d_out;

  char* ws = (char*)d_ws;
  size_t qkv_bytes = (size_t)B_ * OC3 * N_ * sizeof(__bf16);   // 302 MB each
  size_t xb_bytes  = (size_t)B_ * C_ * N_ * sizeof(__bf16);    // 101 MB
  size_t wqb_bytes = (size_t)OC3 * C_ * sizeof(__bf16);        // 0.9 MB
  __bf16* qkv1 = (__bf16*)ws;                                  // pre-dw
  __bf16* qkv2 = (__bf16*)(ws + qkv_bytes);                    // post-dw (q|k|v)
  __bf16* xb   = (__bf16*)(ws + 2 * qkv_bytes);                // bf16 x
  __bf16* wqb  = (__bf16*)(ws + 2 * qkv_bytes + xb_bytes);     // bf16 w_qkv
  float*  inv  = (float*)(ws + 2 * qkv_bytes + xb_bytes + wqb_bytes);
  float*  attn = inv + 1536;                                   // [2][8][48][48]
  __bf16* weff = (__bf16*)(attn + (size_t)B_ * HEADS * HD * HD);

  // 0) one-time bf16 downcasts (x is L2-resident at 101 MB afterwards)
  {
    long n4 = (long)B_ * C_ * N_ / 4;
    cvt_f32_bf16<<<dim3((unsigned)((n4 + 255) / 256)), 256, 0, stream>>>(x, xb, n4);
    long w4 = (long)OC3 * C_ / 4;
    cvt_f32_bf16<<<dim3((unsigned)((w4 + 255) / 256)), 256, 0, stream>>>(w_qkv, wqb, w4);
  }

  // 1) qkv = W_qkv @ x   (per batch: 1152x384 @ 384x65536)
  gemm_wmma<__bf16><<<dim3(512, 9, B_), 256, 0, stream>>>(
      wqb, xb, qkv1, OC3, C_, N_, 0L, (long)C_ * N_, (long)OC3 * N_);

  // 2) depthwise 3x3
  dwconv3x3<<<dim3(H_, OC3, B_), W_, 0, stream>>>(qkv1, w_dw, qkv2);

  // 3) inverse L2 norms of q and k rows
  rownorms<<<dim3(1536), 256, 0, stream>>>(qkv2, inv);

  // 4) attn = softmax(norm(q) @ norm(k)^T * tau)
  chan_attn<<<dim3(B_ * HEADS), 256, 0, stream>>>(qkv2, inv, temperature, attn);

  // 5) W_eff = W_proj @ blockdiag(attn)
  build_weff<<<dim3(C_, B_), 384, 0, stream>>>(w_proj, attn, weff);

  // 6) out = W_eff @ v   (per batch: 384x384 @ 384x65536, f32 out)
  gemm_wmma<float><<<dim3(512, 3, B_), 256, 0, stream>>>(
      weff, qkv2 + (size_t)2 * C_ * N_, out, C_, C_, N_,
      (long)C_ * C_, (long)OC3 * N_, (long)C_ * N_);
}
